// GraphSelfAttentionLayer_5420248727662
// MI455X (gfx1250) — matrix operand order repaired and checked by reading, not compiled, not run on hardware
//
#include <hip/hip_runtime.h>
#include <hip/hip_bf16.h>

// ---------------------------------------------------------------------------
// GraphSelfAttention for MI455X (gfx1250, wave32, WMMA, 320KB LDS/WGP).
//  * y = (att @ V) @ Wout^T  ==  att @ (V @ Wout^T)   (97 -> 10 GFLOP)
//  * scores+softmax+output fused; 311KB score tile lives in LDS (CDNA5-only),
//    eliminating 480 MB of HBM round-trips.
//  * 16x64 output tiles per wave: each A fragment feeds 4 WMMAs.
// ---------------------------------------------------------------------------

typedef __bf16 bf16;
typedef __attribute__((ext_vector_type(16))) __bf16 bf16x16;
typedef __attribute__((ext_vector_type(8)))  __bf16 bf16x8;
typedef __attribute__((ext_vector_type(8)))  float  f32x8;

static constexpr int Bb   = 8;     // batch
static constexpr int Nn   = 1024;  // tokens
static constexpr int Ff   = 1024;  // feat dim
static constexpr int Hh   = 16;    // heads
static constexpr int Dd   = 64;    // head dim
static constexpr int NG   = 300;   // nongt dim
static constexpr int NGP  = 320;   // padded to multiple of 32 for WMMA K loops
static constexpr int MROW = 304;   // per-row f32 slots in LDS score tile

__device__ __forceinline__ bf16x8 load_bf8(const bf16* p) {
    return *reinterpret_cast<const bf16x8*>(p);
}
__device__ __forceinline__ bf16x16 load_bf16(const bf16* p) {
    return *reinterpret_cast<const bf16x16*>(p);
}
__device__ __forceinline__ bf16x16 combine8(bf16x8 lo, bf16x8 hi) {
    bf16x16 r;
#pragma unroll
    for (int i = 0; i < 8; ++i) { r[i] = lo[i]; r[8 + i] = hi[i]; }
    return r;
}
__device__ __forceinline__ f32x8 wmma_bf(bf16x16 a, bf16x16 b, f32x8 c) {
    return __builtin_amdgcn_wmma_f32_16x16x32_bf16(false, a, false, b,
                                                   (short)0, c, false, false);
}

// A fragment: row-major A, lane L holds row (L%16); half-wave contiguous K:
//   lanes 0-15 -> K = k0+[0..15];  lanes 16-31 -> K = k0+[16..31].
// B fragment (NT: B[k][n] = W[n][k], W row-major): lane holds col (L%16),
//   16 contiguous K elements starting at k0 + (lane<16 ? 0 : 16).
// D fragment: VGPR r <-> row (r + 8*(lane>=16)), col (lane%16).

// ---------------------------------------------------------------------------
// K0: f32 -> bf16 conversion
__global__ void f2bf_kernel(const float* __restrict__ x, bf16* __restrict__ y, long n) {
    long i = (long)blockIdx.x * blockDim.x + threadIdx.x;
    long stride = (long)gridDim.x * blockDim.x;
    for (; i < n; i += stride) y[i] = (bf16)x[i];
}

// ---------------------------------------------------------------------------
// K1: C[row, j] = sum_k A[arow,k] * W[j,k] (+ bias[j]), bf16 out.
// One wave computes a 16x64 tile (4 accumulators, A fragment reused 4x).
// mode 0: q    (rows = B*N,   out[row*1024+j])
// mode 1: k    (rows = B*NGP, A-row clamp m<300, out[(b,h,m,dd)])
// mode 2: kvT  (rows = B*NGP, W = Wout flat,     out[(b,h,dd,m)] transposed)
__global__ void gemm_nt_kernel(const bf16* __restrict__ A,
                               const bf16* __restrict__ W,
                               const float* __restrict__ bias,
                               bf16* __restrict__ Out, int mode) {
    const int lane = threadIdx.x & 31;
    const int r16  = lane & 15;
    const int hs   = lane >> 4;
    const int rt   = blockIdx.x;
    const int j0   = blockIdx.y * 64;

    const int orow = rt * 16 + r16;
    long arow;
    if (mode == 0) {
        arow = orow;
    } else {
        int b = orow / NGP, m = orow % NGP;
        int mc = m < NG ? m : NG - 1;      // clamp padded rows (finite values ok)
        arow = (long)b * Nn + mc;
    }
    const bf16* Arow = A + arow * (long)Ff;

    f32x8 acc[4];
#pragma unroll
    for (int jj = 0; jj < 4; ++jj) {
        float binit = bias ? bias[j0 + jj * 16 + r16] : 0.0f;
#pragma unroll
        for (int i = 0; i < 8; ++i) acc[jj][i] = binit;
    }
    const bf16* Wcol = W + (long)(j0 + r16) * Ff + (hs ? 16 : 0);

    for (int k0 = 0; k0 < Ff; k0 += 32) {
        int ak = k0 + (hs ? 16 : 0);
        bf16x16 a = combine8(load_bf8(Arow + ak), load_bf8(Arow + ak + 8));
#pragma unroll
        for (int jj = 0; jj < 4; ++jj) {
            bf16x16 bm = load_bf16(Wcol + (long)jj * 16 * Ff + k0);
            acc[jj] = wmma_bf(a, bm, acc[jj]);
        }
    }

#pragma unroll
    for (int jj = 0; jj < 4; ++jj) {
#pragma unroll
        for (int r = 0; r < 8; ++r) {
            int rowl  = r + 8 * hs;
            int orow2 = rt * 16 + rowl;
            int j     = j0 + jj * 16 + r16;
            bf16 v    = (bf16)acc[jj][r];
            if (mode == 0) {
                Out[(long)orow2 * Ff + j] = v;
            } else {
                int b2 = orow2 / NGP, m2 = orow2 % NGP;
                int h2 = j >> 6, dd = j & 63;
                if (mode == 1)
                    Out[(((long)(b2 * Hh + h2) * NGP) + m2) * Dd + dd] = v;
                else
                    Out[((long)(b2 * Hh + h2) * Dd + dd) * NGP + m2] = v;
            }
        }
    }
}

// ---------------------------------------------------------------------------
// K2: fused scores + softmax + output GEMM. One block per (b, 16-row n-tile),
// 16 waves. The full (16 n) x (16 h) x 300 score tile lives in LDS (311 KB).
//
// Per m-tile (19 iters):
//   phase 1: wave w = n-row: pos-bias tile via WMMA (rows=m, cols=h),
//            apply log(max(relu(.),1e-6)), scatter into scoreS[n=w][h][m].
//   phase 2: wave w = head:  affinity tile via WMMA (rows=n, cols=m),
//            s = aff/8 + posbias, adj mask, + label -> scoreS[n][h=w][m].
// Then in-LDS softmax (wave per (n,h) row), writing bf16 probabilities in
// place over the f32 row; then y = p @ KV with A fragments read from LDS.
__global__ __launch_bounds__(512)
void fused_attn_kernel(const bf16* __restrict__ q,      // [B][N][H*64]
                       const bf16* __restrict__ kk,     // [B][H][NGP][64]
                       const float* __restrict__ posemb,// [B][N][300][64]
                       const bf16* __restrict__ wpos,   // [16][64]
                       const float* __restrict__ bpos,  // [16]
                       const int*   __restrict__ adj,   // [B][N][300]
                       const float* __restrict__ label, // [B][N][300]
                       const bf16* __restrict__ kvT,    // [B][H][64][NGP]
                       const float* __restrict__ bout,  // [1024]
                       float* __restrict__ out) {       // [B][N][1024]
    __shared__ float scoreS[16][16][MROW];  // [n][h][m] : 311,296 B (CDNA5 LDS)

    const int tid  = threadIdx.x;
    const int w    = tid >> 5;   // wave id: n-row in phase1, head in phase2/GEMM
    const int lane = tid & 31;
    const int r16  = lane & 15;
    const int hs   = lane >> 4;
    const int b    = blockIdx.y;
    const int n0   = blockIdx.x * 16;

    // ---- hoisted fragments (read once) ----
    const bf16* qrow = q + ((long)(b * Nn + n0 + r16) * Ff) + w * Dd;
    bf16x16 qf0, qf1;
    {
        int ak = hs ? 16 : 0;
        qf0 = combine8(load_bf8(qrow + ak),      load_bf8(qrow + ak + 8));
        qf1 = combine8(load_bf8(qrow + 32 + ak), load_bf8(qrow + 32 + ak + 8));
    }
    const bf16* wrow = wpos + (long)r16 * 64;   // B col h = lane%16
    bf16x16 wf0 = load_bf16(wrow + (hs ? 16 : 0));
    bf16x16 wf1 = load_bf16(wrow + 32 + (hs ? 16 : 0));
    const float bp = bpos[r16];

    const float* posbase = posemb + ((long)(b * Nn + n0 + w) * NG) * 64;
    const long   kbase   = ((long)(b * Hh + w) * NGP) * Dd;

    for (int mt = 0; mt < 19; ++mt) {
        const int m0 = mt * 16;

        // ---- phase 1: position logits for n = n0 + w ----
        int mA = m0 + r16; if (mA > NG - 1) mA = NG - 1;
        const float* prow = posbase + (long)mA * 64;
        if (mt < 18) __builtin_prefetch(prow + 16 * 64, 0, 1); // next m-tile
        f32x8 accP;
#pragma unroll
        for (int i = 0; i < 8; ++i) accP[i] = bp;
#pragma unroll
        for (int k0 = 0; k0 < 64; k0 += 32) {
            int ak = k0 + (hs ? 16 : 0);
            float4 f0 = *reinterpret_cast<const float4*>(prow + ak);
            float4 f1 = *reinterpret_cast<const float4*>(prow + ak + 4);
            float4 f2 = *reinterpret_cast<const float4*>(prow + ak + 8);
            float4 f3 = *reinterpret_cast<const float4*>(prow + ak + 12);
            bf16x8 lo, hi;
            lo[0]=(bf16)f0.x; lo[1]=(bf16)f0.y; lo[2]=(bf16)f0.z; lo[3]=(bf16)f0.w;
            lo[4]=(bf16)f1.x; lo[5]=(bf16)f1.y; lo[6]=(bf16)f1.z; lo[7]=(bf16)f1.w;
            hi[0]=(bf16)f2.x; hi[1]=(bf16)f2.y; hi[2]=(bf16)f2.z; hi[3]=(bf16)f2.w;
            hi[4]=(bf16)f3.x; hi[5]=(bf16)f3.y; hi[6]=(bf16)f3.z; hi[7]=(bf16)f3.w;
            accP = wmma_bf(combine8(lo, hi), (k0 == 0) ? wf0 : wf1, accP);
        }
#pragma unroll
        for (int r = 0; r < 8; ++r) {
            float lp = __logf(fmaxf(fmaxf(accP[r], 0.0f), 1e-6f));
            scoreS[w][r16][m0 + r + 8 * hs] = lp;   // [n=w][h][m]
        }
        __syncthreads();

        // ---- phase 2: affinity for head h = w, combine ----
        const bf16* krow = kk + kbase + (long)(m0 + r16) * Dd;
        f32x8 accA;
#pragma unroll
        for (int i = 0; i < 8; ++i) accA[i] = 0.0f;
        accA = wmma_bf(qf0, load_bf16(krow + (hs ? 16 : 0)), accA);
        accA = wmma_bf(qf1, load_bf16(krow + 32 + (hs ? 16 : 0)), accA);

        const int m = m0 + r16;   // this lane's m column
        if (m < NG) {
            const int*   adjc = adj   + ((long)(b * Nn + n0) * NG) + m;
            const float* lblc = label + ((long)(b * Nn + n0) * NG) + m;
#pragma unroll
            for (int r = 0; r < 8; ++r) {
                int nl = r + 8 * hs;
                float s = accA[r] * 0.125f + scoreS[nl][w][m];
                s = (adjc[(long)nl * NG] > 0) ? s : -9.0e15f;
                s += lblc[(long)nl * NG];
                scoreS[nl][w][m] = s;
            }
        }
        // no barrier needed: next phase-1 writes a disjoint m range
    }
    __syncthreads();

    // ---- softmax in LDS; wave w owns rows (n=w, h=0..15); p -> bf16 in place
    for (int h = 0; h < 16; ++h) {
        float* srow = &scoreS[w][h][0];
        float vals[10];
        float mx = -3.0e38f;
#pragma unroll
        for (int i = 0; i < 10; ++i) {
            int m = lane + i * 32;
            float v = (m < NG) ? srow[m] : -3.0e38f;
            vals[i] = v;
            mx = fmaxf(mx, v);
        }
#pragma unroll
        for (int off = 16; off; off >>= 1) mx = fmaxf(mx, __shfl_xor(mx, off, 32));
        float sum = 0.0f;
#pragma unroll
        for (int i = 0; i < 10; ++i) {
            int m = lane + i * 32;
            float e = (m < NG) ? __expf(vals[i] - mx) : 0.0f;
            vals[i] = e;
            sum += e;
        }
#pragma unroll
        for (int off = 16; off; off >>= 1) sum += __shfl_xor(sum, off, 32);
        float inv = 1.0f / sum;
        bf16* pb = reinterpret_cast<bf16*>(srow);   // 320 bf16 (640B) < 1216B row
#pragma unroll
        for (int i = 0; i < 10; ++i) pb[lane + i * 32] = (bf16)(vals[i] * inv);
    }
    __syncthreads();

    // ---- output GEMM: y[n, w*64+dd] = p[n, :] @ KV[:, dd] ----
    // k-outer / dd-strip-inner: each LDS A fragment feeds 4 WMMAs.
    const long bh = (long)(b * Hh + w);
    f32x8 acc[4];
#pragma unroll
    for (int dt = 0; dt < 4; ++dt)
#pragma unroll
        for (int i = 0; i < 8; ++i) acc[dt][i] = 0.0f;

    const bf16* arow  = reinterpret_cast<const bf16*>(&scoreS[r16][w][0]);
    const bf16* kcol0 = kvT + (bh * Dd + r16) * (long)NGP + (hs ? 16 : 0);
#pragma unroll
    for (int k0 = 0; k0 < NGP; k0 += 32) {
        int ak = k0 + (hs ? 16 : 0);
        bf16x16 a = combine8(load_bf8(arow + ak), load_bf8(arow + ak + 8));
#pragma unroll
        for (int dt = 0; dt < 4; ++dt) {
            bf16x16 bm = load_bf16(kcol0 + (long)dt * 16 * NGP + k0);
            acc[dt] = wmma_bf(a, bm, acc[dt]);
        }
    }
#pragma unroll
    for (int dt = 0; dt < 4; ++dt) {
        const int f = w * Dd + dt * 16 + r16;
        const float bo = bout[f];
#pragma unroll
        for (int r = 0; r < 8; ++r) {
            int n = n0 + r + 8 * hs;
            out[((long)(b * Nn + n) * Ff) + f] = fmaxf(acc[dt][r] + bo, 0.0f);
        }
    }
}

// ---------------------------------------------------------------------------
extern "C" void kernel_launch(void* const* d_in, const int* in_sizes, int n_in,
                              void* d_out, int out_size, void* d_ws, size_t ws_size,
                              hipStream_t stream) {
    const float* roi   = (const float*)d_in[0];
    const int*   adj   = (const int*)  d_in[1];
    const float* pose  = (const float*)d_in[2];
    const float* label = (const float*)d_in[3];
    const float* Wq    = (const float*)d_in[4];
    const float* bq    = (const float*)d_in[5];
    const float* Wk    = (const float*)d_in[6];
    const float* bk    = (const float*)d_in[7];
    const float* Wpos  = (const float*)d_in[8];
    const float* bpos  = (const float*)d_in[9];
    const float* Wout  = (const float*)d_in[10];
    const float* bout  = (const float*)d_in[11];
    float* out = (float*)d_out;

    // workspace carve (~48 MB)
    char* ws = (char*)d_ws;
    auto carve = [&](size_t bytes) {
        char* r = ws;
        ws += (bytes + 255) & ~(size_t)255;
        return r;
    };
    bf16* roi_bf  = (bf16*)carve((size_t)Bb * Nn * Ff * 2);        // 16 MB
    bf16* wq_bf   = (bf16*)carve((size_t)Ff * Ff * 2);             //  2 MB
    bf16* wk_bf   = (bf16*)carve((size_t)Ff * Ff * 2);             //  2 MB
    bf16* wout_bf = (bf16*)carve((size_t)Ff * Ff * 2);             //  2 MB
    bf16* wpos_bf = (bf16*)carve((size_t)Hh * 64 * 2);
    bf16* q_bf    = (bf16*)carve((size_t)Bb * Nn * Ff * 2);        // 16 MB
    bf16* k_bf    = (bf16*)carve((size_t)Bb * Hh * NGP * Dd * 2);  //  5 MB
    bf16* kvT_bf  = (bf16*)carve((size_t)Bb * Hh * Dd * NGP * 2);  //  5 MB

    // K0: conversions
    f2bf_kernel<<<4096, 256, 0, stream>>>(roi,  roi_bf,  (long)Bb * Nn * Ff);
    f2bf_kernel<<<1024, 256, 0, stream>>>(Wq,   wq_bf,   (long)Ff * Ff);
    f2bf_kernel<<<1024, 256, 0, stream>>>(Wk,   wk_bf,   (long)Ff * Ff);
    f2bf_kernel<<<1024, 256, 0, stream>>>(Wout, wout_bf, (long)Ff * Ff);
    f2bf_kernel<<<4,    256, 0, stream>>>(Wpos, wpos_bf, (long)Hh * 64);

    // K1: projection GEMMs (q, k, KV = V @ Wout^T), 16x64 tiles per wave
    gemm_nt_kernel<<<dim3(Bb * Nn / 16, Ff / 64), 32, 0, stream>>>(
        roi_bf, wq_bf, bq, q_bf, 0);
    gemm_nt_kernel<<<dim3(Bb * NGP / 16, Ff / 64), 32, 0, stream>>>(
        roi_bf, wk_bf, bk, k_bf, 1);
    gemm_nt_kernel<<<dim3(Bb * NGP / 16, Ff / 64), 32, 0, stream>>>(
        roi_bf, wout_bf, nullptr, kvT_bf, 2);

    // K2: fused scores + softmax + output
    fused_attn_kernel<<<dim3(Nn / 16, Bb), 512, 0, stream>>>(
        q_bf, k_bf, pose, wpos_bf, bpos, adj, label, kvT_bf, bout, out);
}
